// TransformerBlock_10505490006735
// MI455X (gfx1250) — matrix operand order repaired and checked by reading, not compiled, not run on hardware
//
#include <hip/hip_runtime.h>
#include <cstddef>

// ---------------------------------------------------------------------------
// Problem constants (match reference)
// ---------------------------------------------------------------------------
constexpr int Bc  = 2;
constexpr int Sc  = 2048;
constexpr int Dc  = 1024;
constexpr int Hc  = 16;
constexpr int Fc  = 4096;
constexpr int HDc = 64;       // Dc / Hc
constexpr int BSc = Bc * Sc;  // 4096 tokens

// ---------------------------------------------------------------------------
// CDNA5 WMMA helpers (bf16 in, f32 accumulate) — v_wmma_f32_16x16x32_bf16
// ---------------------------------------------------------------------------
typedef __attribute__((ext_vector_type(16))) __bf16 v16bf;
typedef __attribute__((ext_vector_type(8)))  __bf16 v8bf;
typedef __attribute__((ext_vector_type(8)))  float  v8f;

// A-layout (16x32, MxK): per lane 16 contiguous elems:
//   elem e -> K = 16*half + e   (at p)
__device__ __forceinline__ v16bf load_a16(const __bf16* p) {
  v8bf lo = *(const v8bf*)(p);
  v8bf hi = *(const v8bf*)(p + 8);
  return __builtin_shufflevector(lo, hi, 0,1,2,3,4,5,6,7,8,9,10,11,12,13,14,15);
}

// B-layout (32x16, KxN): per lane 16 contiguous elems:
//   elem e -> K = 16*half + e   (fixed N = lane%16; source row is Wt[N][.])
__device__ __forceinline__ v16bf load_b16(const __bf16* p) {
  v8bf lo = *(const v8bf*)(p);
  v8bf hi = *(const v8bf*)(p + 8);
  return __builtin_shufflevector(lo, hi, 0,1,2,3,4,5,6,7,8,9,10,11,12,13,14,15);
}

__device__ __forceinline__ v8f wmma_bf16(v16bf a, v16bf b, v8f c) {
  // (neg_a, A, neg_b, B, c_mod, C, reuse_a, reuse_b)
  return __builtin_amdgcn_wmma_f32_16x16x32_bf16(false, a, false, b,
                                                 (short)0, c, false, false);
}

// ---------------------------------------------------------------------------
// Weight convert + transpose:  Wt[n*K + k] = (bf16) W[k*N + n]
// grid: (K/256, N), block 256. Writes coalesced along K.
// ---------------------------------------------------------------------------
__global__ __launch_bounds__(256)
void wconv_kernel(const float* __restrict__ W, __bf16* __restrict__ Wt,
                  int K, int N) {
  int k = blockIdx.x * 256 + threadIdx.x;
  int n = blockIdx.y;
  Wt[(size_t)n * K + k] = (__bf16)W[(size_t)k * N + n];
}

// ---------------------------------------------------------------------------
// LayerNorm (fp32 in -> bf16 out), one row (D=1024) per 256-thread block
// ---------------------------------------------------------------------------
__global__ __launch_bounds__(256)
void ln_kernel(const float* __restrict__ x, const float* __restrict__ g,
               const float* __restrict__ be, __bf16* __restrict__ out) {
  __shared__ float s1[256];
  __shared__ float s2[256];
  const int row = blockIdx.x;
  const float* xr = x + (size_t)row * Dc;
  float a = 0.f, b = 0.f;
  for (int i = threadIdx.x; i < Dc; i += 256) {
    float v = xr[i];
    a += v; b += v * v;
  }
  s1[threadIdx.x] = a; s2[threadIdx.x] = b;
  __syncthreads();
  for (int st = 128; st > 0; st >>= 1) {
    if (threadIdx.x < st) {
      s1[threadIdx.x] += s1[threadIdx.x + st];
      s2[threadIdx.x] += s2[threadIdx.x + st];
    }
    __syncthreads();
  }
  float mean = s1[0] * (1.0f / Dc);
  float var  = s2[0] * (1.0f / Dc) - mean * mean;
  float inv  = rsqrtf(var + 1e-5f);
  __bf16* orow = out + (size_t)row * Dc;
  for (int i = threadIdx.x; i < Dc; i += 256)
    orow[i] = (__bf16)((xr[i] - mean) * inv * g[i] + be[i]);
}

// ---------------------------------------------------------------------------
// V transpose: vT[(bh*HD + hd)*S + s] = v[(b*S+s)*D + h*HD + hd]
// grid: (S/256, HD, B*H)
// ---------------------------------------------------------------------------
__global__ __launch_bounds__(256)
void vtrans_kernel(const __bf16* __restrict__ v, __bf16* __restrict__ vT) {
  int s  = blockIdx.x * 256 + threadIdx.x;
  int hd = blockIdx.y;
  int bh = blockIdx.z;
  int b = bh >> 4, h = bh & 15;
  vT[((size_t)bh * HDc + hd) * Sc + s] =
      v[((size_t)(b * Sc + s)) * Dc + h * HDc + hd];
}

// ---------------------------------------------------------------------------
// Tiled bf16 WMMA GEMM:  out = epi(A[M,K] @ Wt[N,K]^T + bias) (+ res)
// Block = 256 threads = 8 waves (4 M-waves x 2 N-waves); wave tile 32x64
// (2x4 WMMA, A fragments reused across 4 N-tiles); WG tile 128x128.
// 8 WMMA per 12 b128 loads per 32-K step.
// EPI: 0 = none, 1 = exact GELU. HAS_RES adds fp32 residual after epilogue.
// ---------------------------------------------------------------------------
template<int EPI, bool HAS_RES, bool OUT_BF, bool OUT_F>
__global__ __launch_bounds__(256)
void gemm_kernel(const __bf16* __restrict__ A, const __bf16* __restrict__ Wt,
                 const float* __restrict__ bias, const float* __restrict__ res,
                 __bf16* __restrict__ outb, float* __restrict__ outf,
                 int M, int N, int K) {
  const int lane = threadIdx.x & 31;
  const int wave = threadIdx.x >> 5;
  const int half = lane >> 4;
  const int l16  = lane & 15;
  const int m0 = blockIdx.y * 128 + (wave >> 1) * 32;
  const int n0 = blockIdx.x * 128 + (wave & 1)  * 64;

  v8f zf = {0.f,0.f,0.f,0.f,0.f,0.f,0.f,0.f};
  v8f acc[2][4] = {{zf, zf, zf, zf}, {zf, zf, zf, zf}};

  const __bf16* a0 = A  + (size_t)(m0 + l16) * K + half * 16;
  const __bf16* a1 = a0 + (size_t)16 * K;
  const __bf16* b0 = Wt + (size_t)(n0 + l16) * K + half * 16;
  const __bf16* b1 = b0 + (size_t)16 * K;
  const __bf16* b2 = b1 + (size_t)16 * K;
  const __bf16* b3 = b2 + (size_t)16 * K;

  for (int kk = 0; kk < K; kk += 32) {
    __builtin_prefetch(a0 + kk + 256, 0, 0);
    __builtin_prefetch(b0 + kk + 256, 0, 0);
    v16bf am0 = load_a16(a0 + kk);
    v16bf am1 = load_a16(a1 + kk);
    v16bf bn0 = load_b16(b0 + kk);
    v16bf bn1 = load_b16(b1 + kk);
    v16bf bn2 = load_b16(b2 + kk);
    v16bf bn3 = load_b16(b3 + kk);
    acc[0][0] = wmma_bf16(am0, bn0, acc[0][0]);
    acc[0][1] = wmma_bf16(am0, bn1, acc[0][1]);
    acc[0][2] = wmma_bf16(am0, bn2, acc[0][2]);
    acc[0][3] = wmma_bf16(am0, bn3, acc[0][3]);
    acc[1][0] = wmma_bf16(am1, bn0, acc[1][0]);
    acc[1][1] = wmma_bf16(am1, bn1, acc[1][1]);
    acc[1][2] = wmma_bf16(am1, bn2, acc[1][2]);
    acc[1][3] = wmma_bf16(am1, bn3, acc[1][3]);
  }

  #pragma unroll
  for (int mt = 0; mt < 2; ++mt)
    #pragma unroll
    for (int nt = 0; nt < 4; ++nt) {
      const int n = n0 + nt * 16 + l16;
      const float bn = bias[n];
      #pragma unroll
      for (int r = 0; r < 8; ++r) {
        const int m = m0 + mt * 16 + r + 8 * half;
        float v = acc[mt][nt][r] + bn;
        if (EPI == 1)  // exact GELU: 0.5*x*(1+erf(x/sqrt(2)))
          v = 0.5f * v * (1.0f + erff(v * 0.70710678118654752f));
        const size_t idx = (size_t)m * N + n;
        if (HAS_RES) v += res[idx];
        if (OUT_F)  outf[idx] = v;
        if (OUT_BF) outb[idx] = (__bf16)v;
      }
    }
}

// ---------------------------------------------------------------------------
// Causal flash attention, wave32: one wave per (b,h, 16 query rows).
// q,k: [B*S, D] bf16 (head slice h*HD..); vT: [B*H, HD, S] bf16.
// QK^T: 2x2 WMMAs per 32-key step (HD=64 -> two K=32 chunks).
// Online softmax with shfl_xor row reductions (within 16-lane N-groups).
// P goes C-layout -> LDS -> A-layout (s_wait_dscnt 0 between), then AV WMMAs.
// ---------------------------------------------------------------------------
__global__ __launch_bounds__(32)
void attn_kernel(const __bf16* __restrict__ qb, const __bf16* __restrict__ kb,
                 const __bf16* __restrict__ vT, __bf16* __restrict__ ob) {
  __shared__ __attribute__((aligned(16))) __bf16 ldsP[16 * 32];
  const int lane = threadIdx.x;
  const int half = lane >> 4;
  const int l16  = lane & 15;
  const int q0 = blockIdx.x * 16;
  const int bh = blockIdx.y;
  const int b = bh >> 4, h = bh & 15;

  // Q tile in A-layout: two K=32 chunks over HD=64
  const __bf16* qp = qb + ((size_t)(b * Sc + q0 + l16)) * Dc + h * HDc;
  const v16bf aq0 = load_a16(qp + 0 * 32 + half * 16);
  const v16bf aq1 = load_a16(qp + 1 * 32 + half * 16);

  v8f zf = {0.f,0.f,0.f,0.f,0.f,0.f,0.f,0.f};
  v8f oacc[4] = {zf, zf, zf, zf};
  float mrow[8], lrow[8];
  #pragma unroll
  for (int r = 0; r < 8; ++r) { mrow[r] = -1e30f; lrow[r] = 0.f; }

  const int kmax = q0 + 16;  // causal: keys 0 .. q0+15
  for (int k0 = 0; k0 < kmax; k0 += 32) {
    // ---- scores: S = Q K^T * (1/sqrt(HD)), two 16-key column tiles ----
    v8f sc[2];
    #pragma unroll
    for (int nt = 0; nt < 2; ++nt) {
      int krow = k0 + nt * 16 + l16;
      int krd  = krow < Sc ? krow : Sc - 1;  // clamp OOB (masked anyway)
      const __bf16* kp = kb + ((size_t)(b * Sc + krd)) * Dc + h * HDc;
      v16bf bk0 = load_b16(kp + 0 * 32 + half * 16);
      v16bf bk1 = load_b16(kp + 1 * 32 + half * 16);
      v8f c = zf;
      c = wmma_bf16(aq0, bk0, c);
      c = wmma_bf16(aq1, bk1, c);
      sc[nt] = c;
    }
    // ---- online softmax update + stage P to LDS (C-layout) ----
    #pragma unroll
    for (int r = 0; r < 8; ++r) {
      const int qrow = q0 + r + 8 * half;
      float s0 = sc[0][r] * 0.125f;           // 1/sqrt(64)
      float s1 = sc[1][r] * 0.125f;
      if (k0 + l16 > qrow)      s0 = -1e30f;  // causal mask
      if (k0 + 16 + l16 > qrow) s1 = -1e30f;
      float mx = fmaxf(s0, s1);
      #pragma unroll
      for (int off = 8; off > 0; off >>= 1)
        mx = fmaxf(mx, __shfl_xor(mx, off, 32));
      const float mnew  = fmaxf(mrow[r], mx);
      const float alpha = __expf(mrow[r] - mnew);
      const float p0 = __expf(s0 - mnew);
      const float p1 = __expf(s1 - mnew);
      float ps = p0 + p1;
      #pragma unroll
      for (int off = 8; off > 0; off >>= 1)
        ps += __shfl_xor(ps, off, 32);
      lrow[r] = lrow[r] * alpha + ps;
      mrow[r] = mnew;
      #pragma unroll
      for (int c4 = 0; c4 < 4; ++c4) oacc[c4][r] *= alpha;
      const int m = r + 8 * half;
      ldsP[m * 32 +      l16] = (__bf16)p0;
      ldsP[m * 32 + 16 + l16] = (__bf16)p1;
    }
    // cross-lane LDS RAW within the wave: drain DS counter
    asm volatile("s_wait_dscnt 0" ::: "memory");
    // re-read P in A-layout (16x32, K=32 keys)
    const v16bf ap = load_a16(&ldsP[l16 * 32 + half * 16]);
    // ---- O += P @ V : 4 hd chunks of 16 ----
    #pragma unroll
    for (int c4 = 0; c4 < 4; ++c4) {
      const __bf16* vp =
          vT + ((size_t)bh * HDc + c4 * 16 + l16) * Sc + k0 + half * 16;
      v16bf bv = load_b16(vp);
      oacc[c4] = wmma_bf16(ap, bv, oacc[c4]);
    }
  }
  // ---- normalize and write bf16 head slice ----
  #pragma unroll
  for (int r = 0; r < 8; ++r) {
    const float inv = 1.0f / lrow[r];
    const int qrow = q0 + r + 8 * half;
    __bf16* op = ob + ((size_t)(b * Sc + qrow)) * Dc + h * HDc;
    #pragma unroll
    for (int c4 = 0; c4 < 4; ++c4)
      op[c4 * 16 + l16] = (__bf16)(oacc[c4][r] * inv);
  }
}

// ---------------------------------------------------------------------------
// Launcher
// ---------------------------------------------------------------------------
extern "C" void kernel_launch(void* const* d_in, const int* in_sizes, int n_in,
                              void* d_out, int out_size, void* d_ws, size_t ws_size,
                              hipStream_t stream) {
  const float* x   = (const float*)d_in[0];
  const float* Wq  = (const float*)d_in[1];
  const float* bq  = (const float*)d_in[2];
  const float* Wk  = (const float*)d_in[3];
  const float* bk  = (const float*)d_in[4];
  const float* Wv  = (const float*)d_in[5];
  const float* bv  = (const float*)d_in[6];
  const float* Wp  = (const float*)d_in[7];
  const float* bp  = (const float*)d_in[8];
  const float* g1  = (const float*)d_in[9];
  const float* be1 = (const float*)d_in[10];
  const float* g2  = (const float*)d_in[11];
  const float* be2 = (const float*)d_in[12];
  const float* W1  = (const float*)d_in[13];
  const float* bf1 = (const float*)d_in[14];
  const float* W2  = (const float*)d_in[15];
  const float* bf2 = (const float*)d_in[16];

  char* ws = (char*)d_ws;
  size_t off = 0;
  auto alloc = [&](size_t bytes) {
    size_t o = off; off += (bytes + 255) & ~(size_t)255; return o;
  };
  const size_t tokD = (size_t)BSc * Dc;      // 4M elements
  const size_t o_wqt = alloc((size_t)Dc * Dc * 2);
  const size_t o_wkt = alloc((size_t)Dc * Dc * 2);
  const size_t o_wvt = alloc((size_t)Dc * Dc * 2);
  const size_t o_wpt = alloc((size_t)Dc * Dc * 2);
  const size_t o_w1t = alloc((size_t)Dc * Fc * 2);
  const size_t o_w2t = alloc((size_t)Fc * Dc * 2);
  const size_t o_ln  = alloc(tokD * 2);            // a_in; reused as f_in
  const size_t o_qkv = alloc(tokD * 2 * 4 + 256);  // q,k,v,vT; reused as h
  const size_t o_atn = alloc(tokD * 2);
  const size_t o_x1  = alloc(tokD * 4);

  __bf16* wqt  = (__bf16*)(ws + o_wqt);
  __bf16* wkt  = (__bf16*)(ws + o_wkt);
  __bf16* wvt  = (__bf16*)(ws + o_wvt);
  __bf16* wpt  = (__bf16*)(ws + o_wpt);
  __bf16* w1t  = (__bf16*)(ws + o_w1t);
  __bf16* w2t  = (__bf16*)(ws + o_w2t);
  __bf16* lnb  = (__bf16*)(ws + o_ln);     // a_in, then f_in
  __bf16* qbuf = (__bf16*)(ws + o_qkv);
  __bf16* kbuf = qbuf + tokD;
  __bf16* vbuf = kbuf + tokD;
  __bf16* vTb  = vbuf + tokD;
  __bf16* hbuf = (__bf16*)(ws + o_qkv);    // reuse q/k/v/vT block (32 MB)
  __bf16* atnb = (__bf16*)(ws + o_atn);
  float*  x1   = (float*)(ws + o_x1);
  float*  outp = (float*)d_out;

  // 1) weight convert + transpose (fp32 [K,N] -> bf16 [N,K])
  wconv_kernel<<<dim3(Dc/256, Dc), 256, 0, stream>>>(Wq, wqt, Dc, Dc);
  wconv_kernel<<<dim3(Dc/256, Dc), 256, 0, stream>>>(Wk, wkt, Dc, Dc);
  wconv_kernel<<<dim3(Dc/256, Dc), 256, 0, stream>>>(Wv, wvt, Dc, Dc);
  wconv_kernel<<<dim3(Dc/256, Dc), 256, 0, stream>>>(Wp, wpt, Dc, Dc);
  wconv_kernel<<<dim3(Dc/256, Fc), 256, 0, stream>>>(W1, w1t, Dc, Fc);
  wconv_kernel<<<dim3(Fc/256, Dc), 256, 0, stream>>>(W2, w2t, Fc, Dc);

  // 2) pre-LN 1
  ln_kernel<<<BSc, 256, 0, stream>>>(x, g1, be1, lnb);

  // 3) Q/K/V projections (bf16 out)
  gemm_kernel<0,false,true,false><<<dim3(Dc/128, BSc/128), 256, 0, stream>>>(
      lnb, wqt, bq, nullptr, qbuf, nullptr, BSc, Dc, Dc);
  gemm_kernel<0,false,true,false><<<dim3(Dc/128, BSc/128), 256, 0, stream>>>(
      lnb, wkt, bk, nullptr, kbuf, nullptr, BSc, Dc, Dc);
  gemm_kernel<0,false,true,false><<<dim3(Dc/128, BSc/128), 256, 0, stream>>>(
      lnb, wvt, bv, nullptr, vbuf, nullptr, BSc, Dc, Dc);

  // 4) V -> V^T [B*H, HD, S] for B-layout AV reads
  vtrans_kernel<<<dim3(Sc/256, HDc, Bc*Hc), 256, 0, stream>>>(vbuf, vTb);

  // 5) causal flash attention
  attn_kernel<<<dim3(Sc/16, Bc*Hc), 32, 0, stream>>>(qbuf, kbuf, vTb, atnb);

  // 6) output projection + residual -> x1 (fp32)
  gemm_kernel<0,true,false,true><<<dim3(Dc/128, BSc/128), 256, 0, stream>>>(
      atnb, wpt, bp, x, nullptr, x1, BSc, Dc, Dc);

  // 7) pre-LN 2
  ln_kernel<<<BSc, 256, 0, stream>>>(x1, g2, be2, lnb);

  // 8) FFN up + exact GELU (bf16 out)
  gemm_kernel<1,false,true,false><<<dim3(Fc/128, BSc/128), 256, 0, stream>>>(
      lnb, w1t, bf1, nullptr, hbuf, nullptr, BSc, Fc, Dc);

  // 9) FFN down + residual -> final fp32 output
  gemm_kernel<0,true,false,true><<<dim3(Dc/128, BSc/128), 256, 0, stream>>>(
      hbuf, w2t, bf2, x1, nullptr, outp, BSc, Dc, Fc);
}
